// MultiScaleResidualQuantizer_19396072309110
// MI455X (gfx1250) — matrix operand order repaired and checked by reading, not compiled, not run on hardware
//
#include <hip/hip_runtime.h>
#include <hip/hip_bf16.h>

typedef __attribute__((ext_vector_type(16))) _Float16 v16h;
typedef __attribute__((ext_vector_type(8)))  _Float16 v8h;
typedef __attribute__((ext_vector_type(8)))  float    v8f;

#define B_    64
#define C_    32
#define H_    16
#define W_    16
#define NE_   4096
#define NPIX  (B_*C_*H_*W_)   /* 524288 */
#define IDX_TOT 341           /* 1+4+16+64+256 */
#define TPW   4               /* 16-row tiles per wave (64 rows/wave) */

// ---------------------------------------------------------------------------
// init: copy z -> z_res, zero z_hat accumulator (lives in d_out) and loss slot
// ---------------------------------------------------------------------------
__global__ void k_init(const float* __restrict__ z, float* __restrict__ z_res,
                       float* __restrict__ z_hat, float* __restrict__ loss)
{
  int i = blockIdx.x * blockDim.x + threadIdx.x;
  if (i < NPIX) { z_res[i] = z[i]; z_hat[i] = 0.0f; }
  if (i == 0) *loss = 0.0f;
}

// ---------------------------------------------------------------------------
// codebook prep: f32 -> f16 conversion + per-row squared norm
// ---------------------------------------------------------------------------
__global__ void k_prep(const float* __restrict__ emb, _Float16* __restrict__ embh,
                       float* __restrict__ embsq)
{
  int r = blockIdx.x * blockDim.x + threadIdx.x;
  if (r >= NE_) return;
  float s = 0.0f;
  #pragma unroll 8
  for (int c = 0; c < C_; ++c) {
    float v = emb[r * C_ + c];
    s += v * v;
    embh[r * C_ + c] = (_Float16)v;
  }
  embsq[r] = s;
}

// ---------------------------------------------------------------------------
// area downsample of residual -> channel-last f16 rows rf[N][32]
// (ph == 16 -> f == 1 -> pure conversion, matching reference's si==K-1 branch)
// ---------------------------------------------------------------------------
__global__ void k_down(const float* __restrict__ z_res, _Float16* __restrict__ rf,
                       int ph)
{
  int f = H_ / ph;
  int N = B_ * ph * ph;
  int i = blockIdx.x * blockDim.x + threadIdx.x;   // over N*C
  if (i >= N * C_) return;
  int c = i & (C_ - 1);
  int n = i >> 5;
  int b = n / (ph * ph);
  int rem = n - b * ph * ph;
  int y = rem / ph, x = rem - (rem / ph) * ph;
  const float* base = z_res + ((b * C_ + c) * H_ + y * f) * W_ + x * f;
  float s = 0.0f;
  for (int dy = 0; dy < f; ++dy)
    for (int dx = 0; dx < f; ++dx)
      s += base[dy * W_ + dx];
  rf[n * C_ + c] = (_Float16)(s / (float)(f * f));
}

// ---------------------------------------------------------------------------
// WMMA nearest-codebook search, register-blocked on rows.
// One wave owns TPW=4 16-row tiles (64 rows) and scans the codebook once:
// per 16-entry codebook tile -> one B-fragment load feeds 4 independent
// v_wmma_f32_16x16x32_f16 (K=32 == C exactly), each with its OWN accumulator
// register set so the back-to-back WMMA issues fill each other's WMMA->VALU
// hazard slots (no v_nop padding); all D consumption happens afterwards.
//
// Fragment layouts per CDNA5 ISA 7.12.2 (wave32):
//   A (16x32 f16): lanes 0-15 = rows, K 0..7 & 16..23; lanes 16-31 K 8..15 & 24..31
//   B (32x16 f16): lane holds K 8*hi + 0..7 & 16 + 8*hi + 0..7 of codebook column n = lo
//   D (16x16 f32): slot v of lane -> row M = v + 8*hi, col N = lo
// dist(m,n) = ||e_n||^2 - 2*dot(r_m,e_n); per-(lane,slot) running argmin, then
// width-16 butterfly reduction across the lanes sharing each row.
// ---------------------------------------------------------------------------
__global__ __launch_bounds__(256) void k_quant(const _Float16* __restrict__ rf,
    const _Float16* __restrict__ embh, const float* __restrict__ embsq,
    int* __restrict__ idx, int numTiles)
{
  int wave  = threadIdx.x >> 5;
  int lane  = threadIdx.x & 31;
  int group = blockIdx.x * 8 + wave;            // group of TPW tiles
  if (group * TPW >= numTiles) return;          // wave-uniform: EXEC stays all-ones
  int hi = lane >> 4, lo = lane & 15;

  v16h a[TPW];
  #pragma unroll
  for (int tI = 0; tI < TPW; ++tI) {
    const _Float16* s = rf + ((group * TPW + tI) * 16 + lo) * C_ + hi * 8;
    v8h p0 = *(const v8h*)(s);                  // K = 8*hi + 0..7
    v8h p1 = *(const v8h*)(s + 16);             // K = 16 + 8*hi + 0..7
    #pragma unroll
    for (int i = 0; i < 8; ++i) { a[tI][i] = p0[i]; a[tI][8 + i] = p1[i]; }
  }

  float bestd[TPW][8];
  int   besti[TPW][8];
  #pragma unroll
  for (int tI = 0; tI < TPW; ++tI)
    #pragma unroll
    for (int v = 0; v < 8; ++v) { bestd[tI][v] = 3.0e38f; besti[tI][v] = 0; }

  __builtin_prefetch(embh + lane * 64, 0, 0);   // global_prefetch_b8 (one-shot)

  for (int t = 0; t < NE_; t += 16) {
    int n = t + lo;
    const _Float16* bs = embh + n * C_ + hi * 8;
    v8h q0 = *(const v8h*)(bs);
    v8h q1 = *(const v8h*)(bs + 16);
    v16h b;
    #pragma unroll
    for (int i = 0; i < 8; ++i) { b[i] = q0[i]; b[8 + i] = q1[i]; }
    float esq = embsq[n];

    // issue phase: 4 independent WMMAs (separate accumulators)
    v8f c[TPW];
    #pragma unroll
    for (int tI = 0; tI < TPW; ++tI) {
      v8f z0 = {};
      c[tI] = __builtin_amdgcn_wmma_f32_16x16x32_f16(
          /*neg_a=*/false, a[tI], /*neg_b=*/false, b,
          /*c_mod=*/(short)0, z0, /*reuse_a=*/false, /*reuse_b=*/false);
    }

    // consume phase: argmin bookkeeping (packs into v_pk_fma_f32 + cndmask)
    #pragma unroll
    for (int tI = 0; tI < TPW; ++tI) {
      #pragma unroll
      for (int v = 0; v < 8; ++v) {
        float d = esq - 2.0f * c[tI][v];
        if (d < bestd[tI][v]) { bestd[tI][v] = d; besti[tI][v] = n; }
      }
    }
  }

  // reduce argmin across the 16 lanes sharing each row (width-16 butterflies)
  #pragma unroll
  for (int tI = 0; tI < TPW; ++tI) {
    #pragma unroll
    for (int v = 0; v < 8; ++v) {
      #pragma unroll
      for (int off = 8; off >= 1; off >>= 1) {
        float od = __shfl_xor(bestd[tI][v], off, 16);
        int   oi = __shfl_xor(besti[tI][v], off, 16);
        if (od < bestd[tI][v] || (od == bestd[tI][v] && oi < besti[tI][v])) {
          bestd[tI][v] = od; besti[tI][v] = oi;
        }
      }
    }
    if (lo == 0) {
      #pragma unroll
      for (int v = 0; v < 8; ++v)
        idx[(group * TPW + tI) * 16 + v + 8 * hi] = besti[tI][v];
    }
  }
}

// ---------------------------------------------------------------------------
// gather quantized vectors (f32, NCHW) + emit index output (as float)
// ---------------------------------------------------------------------------
__global__ void k_gather(const int* __restrict__ idx, const float* __restrict__ emb,
                         float* __restrict__ q_small, float* __restrict__ idx_out,
                         int ph, int scale_off)
{
  int NPh = B_ * ph * ph;
  int i = blockIdx.x * blockDim.x + threadIdx.x;   // over NPh*C
  if (i >= NPh * C_) return;
  int c = i & 31;
  int n = i >> 5;
  int b = n / (ph * ph);
  int rem = n - b * ph * ph;
  int id = idx[n];
  q_small[(b * C_ + c) * ph * ph + rem] = emb[id * C_ + c];
  if (c == 0) idx_out[b * IDX_TOT + scale_off + rem] = (float)id;
}

// ---------------------------------------------------------------------------
// bicubic upsample (PyTorch kernel, A = -0.75, align_corners=False)
// ---------------------------------------------------------------------------
__device__ __forceinline__ float cubicw(float t)
{
  const float A = -0.75f;
  t = fabsf(t);
  if (t <= 1.0f) return ((A + 2.0f) * t - (A + 3.0f)) * t * t + 1.0f;
  if (t < 2.0f)  return A * (((t - 5.0f) * t + 8.0f) * t - 4.0f);
  return 0.0f;
}

__global__ void k_bicubic(const float* __restrict__ qs, float* __restrict__ qu, int ph)
{
  int i = blockIdx.x * blockDim.x + threadIdx.x;   // over B*C*16*16
  if (i >= NPIX) return;
  int x = i & 15, y = (i >> 4) & 15, bc = i >> 8;
  float scale = (float)ph / 16.0f;
  float sy = ((float)y + 0.5f) * scale - 0.5f;
  float sx = ((float)x + 0.5f) * scale - 0.5f;
  int iy = (int)floorf(sy), ix = (int)floorf(sx);
  float fy = sy - (float)iy, fx = sx - (float)ix;
  float wy[4], wx[4];
  int yy[4], xx[4];
  #pragma unroll
  for (int j = 0; j < 4; ++j) {
    wy[j] = cubicw(fy - (float)(j - 1));
    wx[j] = cubicw(fx - (float)(j - 1));
    int t = iy + j - 1; yy[j] = t < 0 ? 0 : (t > ph - 1 ? ph - 1 : t);
    t     = ix + j - 1; xx[j] = t < 0 ? 0 : (t > ph - 1 ? ph - 1 : t);
  }
  const float* base = qs + bc * ph * ph;
  float acc = 0.0f;
  #pragma unroll
  for (int j = 0; j < 4; ++j) {
    float r = 0.0f;
    #pragma unroll
    for (int k = 0; k < 4; ++k) r += base[yy[j] * ph + xx[k]] * wx[k];
    acc += r * wy[j];
  }
  qu[i] = acc;
}

// ---------------------------------------------------------------------------
// Phi + state update + loss: q = 0.5*x + 0.5*(conv3x3(x)+b);
// z_hat += q; z_res -= q; loss += lscale * sum((z_hat - z)^2).
// Block per batch image; input tile staged in LDS (32 KB), weights are
// wave-uniform -> scalar-cached reads.
// ---------------------------------------------------------------------------
__global__ __launch_bounds__(256) void k_phi(const float* __restrict__ qu,
    const float* __restrict__ w, const float* __restrict__ bias,
    const float* __restrict__ z, float* __restrict__ z_hat,
    float* __restrict__ z_res, float* __restrict__ loss, float lscale)
{
  __shared__ float sIn[C_ * H_ * W_];   // 8192 floats
  __shared__ float sRed[256];
  int b = blockIdx.x;
  int tid = threadIdx.x;
  const float* inB = qu + b * C_ * H_ * W_;
  for (int i = tid; i < C_ * H_ * W_; i += 256) sIn[i] = inB[i];
  __syncthreads();

  int y = tid >> 4, x = tid & 15;
  float lsum = 0.0f;
  for (int co = 0; co < C_; ++co) {
    float acc = 0.0f;
    const float* wc = w + co * C_ * 9;
    for (int ci = 0; ci < C_; ++ci) {
      const float* in = sIn + ci * 256;
      const float* wk = wc + ci * 9;
      #pragma unroll
      for (int ky = 0; ky < 3; ++ky) {
        int iy = y + ky - 1;
        if ((unsigned)iy >= 16u) continue;
        #pragma unroll
        for (int kx = 0; kx < 3; ++kx) {
          int ix = x + kx - 1;
          if ((unsigned)ix >= 16u) continue;
          acc += in[iy * 16 + ix] * wk[ky * 3 + kx];
        }
      }
    }
    float qv = 0.5f * sIn[co * 256 + (y << 4) + x] + 0.5f * (acc + bias[co]);
    int g = ((b * C_ + co) * H_ + y) * W_ + x;
    float zh = z_hat[g] + qv;
    z_hat[g] = zh;
    z_res[g] -= qv;
    float d = zh - z[g];
    lsum += d * d;
  }
  sRed[tid] = lsum;
  __syncthreads();
  for (int s = 128; s; s >>= 1) {
    if (tid < s) sRed[tid] += sRed[tid + s];
    __syncthreads();
  }
  if (tid == 0) atomicAdd(loss, sRed[0] * lscale);
}

// ---------------------------------------------------------------------------
extern "C" void kernel_launch(void* const* d_in, const int* in_sizes, int n_in,
                              void* d_out, int out_size, void* d_ws, size_t ws_size,
                              hipStream_t stream)
{
  const float* z   = (const float*)d_in[0];   // [64,32,16,16]
  const float* emb = (const float*)d_in[1];   // [4096,32]
  const float* phw = (const float*)d_in[2];   // [4,32,32,3,3]
  const float* phb = (const float*)d_in[3];   // [4,32]

  float* out_zhat = (float*)d_out;            // 524288
  float* out_loss = out_zhat + NPIX;          // 1
  float* out_idx  = out_loss + 1;             // 64*341 (as float)

  char* ws = (char*)d_ws;
  size_t off = 0;
  auto walloc = [&](size_t bytes) -> void* {
    void* p = ws + off;
    off = (off + bytes + 255) & ~(size_t)255;
    return p;
  };
  float*    z_res = (float*)    walloc((size_t)NPIX * 4);
  _Float16* embh  = (_Float16*) walloc((size_t)NE_ * C_ * 2);
  float*    embsq = (float*)    walloc((size_t)NE_ * 4);
  _Float16* rf    = (_Float16*) walloc((size_t)B_ * 256 * C_ * 2);
  int*      idxb  = (int*)      walloc((size_t)B_ * 256 * 4);
  float*    qs    = (float*)    walloc((size_t)NPIX * 4);
  float*    qu    = (float*)    walloc((size_t)NPIX * 4);
  (void)ws_size; (void)in_sizes; (void)n_in; (void)out_size;

  k_init<<<(NPIX + 255) / 256, 256, 0, stream>>>(z, z_res, out_zhat, out_loss);
  k_prep<<<(NE_ + 255) / 256, 256, 0, stream>>>(emb, embh, embsq);

  const int scales[5] = {1, 2, 4, 8, 16};
  const int pisel[5]  = {0, 1, 1, 2, 3};    // static Phi pick from ticks
  const int soff[5]   = {0, 1, 5, 21, 85};  // cumulative per-b index offsets
  const float lscale  = (1.0f + 0.25f) / (5.0f * (float)NPIX);

  for (int si = 0; si < 5; ++si) {
    int ph = scales[si];
    int N  = B_ * ph * ph;
    k_down<<<(N * C_ + 255) / 256, 256, 0, stream>>>(z_res, rf, ph);
    int tiles = N / 16;                       // N is a multiple of 64 = 16*TPW
    int waves = tiles / TPW;
    k_quant<<<(waves + 7) / 8, 256, 0, stream>>>(rf, embh, embsq, idxb, tiles);
    k_gather<<<(N * C_ + 255) / 256, 256, 0, stream>>>(idxb, emb, qs, out_idx, ph, soff[si]);
    const float* qin = qs;
    if (si != 4) {
      k_bicubic<<<(NPIX + 255) / 256, 256, 0, stream>>>(qs, qu, ph);
      qin = qu;
    }
    k_phi<<<B_, 256, 0, stream>>>(qin, phw + (size_t)pisel[si] * C_ * C_ * 9,
                                  phb + (size_t)pisel[si] * C_, z,
                                  out_zhat, z_res, out_loss, lscale);
  }
}
